// LinkPredictor_89644557402729
// MI455X (gfx1250) — matrix operand order repaired and checked by reading, not compiled, not run on hardware
//
#include <hip/hip_runtime.h>
#include <hip/hip_bf16.h>
#include <math.h>

typedef __attribute__((ext_vector_type(16))) _Float16 v16h;
typedef __attribute__((ext_vector_type(8)))  _Float16 v8h;
typedef __attribute__((ext_vector_type(8)))  float    v8f;

#define IN_DIM 128   // per-endpoint feature dim; concat -> 256
#define K1 256
#define HID 64

// Wave-internal LDS ordering: DS ops from one wave are in-order at the LDS,
// but keep the compiler from reordering and make store->load visible.
__device__ __forceinline__ void wave_lds_fence() {
    __builtin_amdgcn_wave_barrier();
    asm volatile("s_wait_dscnt 0x0" ::: "memory");
    __builtin_amdgcn_wave_barrier();
}

__global__ __launch_bounds__(256) void linkpred_mlp_kernel(
    const float* __restrict__ z,
    const long long* __restrict__ edge,   // [2, E] int64
    const float* __restrict__ W1, const float* __restrict__ b1,
    const float* __restrict__ W2, const float* __restrict__ b2,
    const float* __restrict__ W3, const float* __restrict__ b3,
    float* __restrict__ out, int E)
{
    // W1^T as f16 [n][k] (32KB), W2^T as f16 [n][k] (4KB), small params,
    // plus 2KB per-wave reshape scratch (f16 16x64 for L1, f32 16x32 for L2).
    __shared__ _Float16 sW1t[HID * K1];
    __shared__ _Float16 sW2t[32 * HID];
    __shared__ float sB1[HID];
    __shared__ float sB2[32];
    __shared__ float sW3[32];
    __shared__ float sB3;
    __shared__ __align__(16) _Float16 sScr[8][1024];

    const int tid = threadIdx.x;

    // ---- stage weights into LDS (fp32 -> f16, transposed) ----
    for (int i = tid; i < K1 * HID; i += 256) {
        int k = i >> 6, n = i & 63;                 // W1[k][n], row-major [256,64]
        sW1t[n * K1 + k] = (_Float16)W1[i];
    }
    for (int i = tid; i < HID * 32; i += 256) {
        int k = i >> 5, n = i & 31;                 // W2[k][n], row-major [64,32]
        sW2t[n * HID + k] = (_Float16)W2[i];
    }
    if (tid < HID) sB1[tid] = b1[tid];
    if (tid < 32)  { sB2[tid] = b2[tid]; sW3[tid] = W3[tid]; }
    if (tid == 0)  sB3 = b3[0];
    __syncthreads();

    const int lane  = tid & 31;
    const int wave  = tid >> 5;
    const int nl    = lane & 15;     // N (or M) index within 16
    const int half  = lane >> 4;     // which half of the wave
    const long long* ei = edge;
    const long long* ej = edge + E;
    const int nTiles  = (E + 15) >> 4;
    const int gwave   = blockIdx.x * 8 + wave;
    const int strideW = gridDim.x * 8;

    _Float16* h1 = sScr[wave];
    float*    h2 = (float*)sScr[wave];

    for (int tile = gwave; tile < nTiles; tile += strideW) {
        int em = tile * 16 + nl;
        if (em >= E) em = E - 1;     // clamp (store is guarded below)
        const float* rowi = z + (size_t)ei[em] * IN_DIM;
        const float* rowj = z + (size_t)ej[em] * IN_DIM;

        // prefetch next tile's edge indices into cache
        {
            int nxt = tile + strideW;
            if (nxt < nTiles) {
                __builtin_prefetch(&ei[nxt * 16 + nl], 0, 1);
                __builtin_prefetch(&ej[nxt * 16 + nl], 0, 1);
            }
        }

        // ================= Layer 1: [16x256] @ [256x64] =================
        v8f acc[4] = {};
        #pragma unroll
        for (int kb = 0; kb < 8; ++kb) {
            // A fragment (16-bit A 16x32 layout): lane half owns
            // K = {h*8 .. h*8+7} in v[0:3] and {16+h*8 ..} in v[4:7].
            const float* src = (kb < 4) ? rowi : rowj;
            const int base = (kb < 4) ? kb * 32 : kb * 32 - 128;
            const int c0 = base + half * 8;
            const int c1 = base + 16 + half * 8;
            v8f f0 = *(const v8f*)(src + c0);
            v8f f1 = *(const v8f*)(src + c1);
            v8h a0 = __builtin_convertvector(f0, v8h);
            v8h a1 = __builtin_convertvector(f1, v8h);
            v16h a = __builtin_shufflevector(a0, a1,
                0,1,2,3,4,5,6,7,8,9,10,11,12,13,14,15);

            #pragma unroll
            for (int nt = 0; nt < 4; ++nt) {
                // B fragment: lane = column n, half owns K = {h*8 .. h*8+7} and {16+h*8 ..}
                const _Float16* wr = sW1t + (nt * 16 + nl) * K1 + kb * 32 + half * 8;
                v8h blo = *(const v8h*)wr;
                v8h bhi = *(const v8h*)(wr + 16);
                v16h b = __builtin_shufflevector(blo, bhi,
                    0,1,2,3,4,5,6,7,8,9,10,11,12,13,14,15);
                acc[nt] = __builtin_amdgcn_wmma_f32_16x16x32_f16(
                    false, a, false, b, (short)0, acc[nt], false, false);
            }
        }

        // bias + ReLU, reshape C layout -> [16 rows][64 cols] f16 in LDS
        #pragma unroll
        for (int nt = 0; nt < 4; ++nt) {
            float bv = sB1[nt * 16 + nl];
            #pragma unroll
            for (int r = 0; r < 8; ++r) {
                float v = acc[nt][r] + bv;
                v = v > 0.f ? v : 0.f;
                h1[(r + 8 * half) * HID + nt * 16 + nl] = (_Float16)v;
            }
        }
        wave_lds_fence();

        // ================= Layer 2: [16x64] @ [64x32] =================
        v8f acc2[2] = {};
        #pragma unroll
        for (int kb = 0; kb < 2; ++kb) {
            const _Float16* hr = h1 + nl * HID + kb * 32;
            v8h a0 = *(const v8h*)(hr + half * 8);
            v8h a1 = *(const v8h*)(hr + 16 + half * 8);
            v16h a = __builtin_shufflevector(a0, a1,
                0,1,2,3,4,5,6,7,8,9,10,11,12,13,14,15);
            #pragma unroll
            for (int nt = 0; nt < 2; ++nt) {
                const _Float16* wr = sW2t + (nt * 16 + nl) * HID + kb * 32 + half * 8;
                v8h blo = *(const v8h*)wr;
                v8h bhi = *(const v8h*)(wr + 16);
                v16h b = __builtin_shufflevector(blo, bhi,
                    0,1,2,3,4,5,6,7,8,9,10,11,12,13,14,15);
                acc2[nt] = __builtin_amdgcn_wmma_f32_16x16x32_f16(
                    false, a, false, b, (short)0, acc2[nt], false, false);
            }
        }
        wave_lds_fence();

        // bias + ReLU -> f32 [16 rows][32 cols] in same per-wave scratch
        #pragma unroll
        for (int nt = 0; nt < 2; ++nt) {
            float bv = sB2[nt * 16 + nl];
            #pragma unroll
            for (int r = 0; r < 8; ++r) {
                float v = acc2[nt][r] + bv;
                v = v > 0.f ? v : 0.f;
                h2[(r + 8 * half) * 32 + nt * 16 + nl] = v;
            }
        }
        wave_lds_fence();

        // ================= Layer 3: [16x32] @ [32x1] + sigmoid ========
        if (lane < 16) {
            int e = tile * 16 + lane;
            if (e < E) {
                float s = sB3;
                const float* hr = h2 + lane * 32;
                #pragma unroll
                for (int k = 0; k < 32; ++k)
                    s = fmaf(hr[k], sW3[k], s);
                out[e] = 1.f / (1.f + __expf(-s));
            }
        }
        wave_lds_fence();   // scratch reused next iteration
    }
}

extern "C" void kernel_launch(void* const* d_in, const int* in_sizes, int n_in,
                              void* d_out, int out_size, void* d_ws, size_t ws_size,
                              hipStream_t stream) {
    const float*     z    = (const float*)d_in[0];
    const long long* edge = (const long long*)d_in[1];   // int64 [2, E]
    const float*     W1   = (const float*)d_in[2];
    const float*     b1   = (const float*)d_in[3];
    const float*     W2   = (const float*)d_in[4];
    const float*     b2   = (const float*)d_in[5];
    const float*     W3   = (const float*)d_in[6];
    const float*     b3   = (const float*)d_in[7];
    float* out = (float*)d_out;

    const int E = in_sizes[1] / 2;
    const int nTiles = (E + 15) / 16;
    int nBlocks = (nTiles + 7) / 8;
    if (nBlocks > 1024) nBlocks = 1024;
    if (nBlocks < 1) nBlocks = 1;

    linkpred_mlp_kernel<<<nBlocks, 256, 0, stream>>>(
        z, edge, W1, b1, W2, b2, W3, b3, out, E);
}
